// TransformerBlock_335007449336
// MI455X (gfx1250) — matrix operand order repaired and checked by reading, compiled, not run on hardware
//
#include <hip/hip_runtime.h>
#include <hip/hip_bf16.h>
#include <math.h>

typedef __attribute__((ext_vector_type(16))) __bf16 bf16x16;
typedef __attribute__((ext_vector_type(8)))  __bf16 bf16x8;
typedef __attribute__((ext_vector_type(8)))  float  f32x8;

constexpr int Bb = 4, Ss = 512, Dd = 4096, Hh = 32, KVh = 8, HD = 128;
constexpr int HID = 14336, ALn = 10, RLn = 64;

__device__ __forceinline__ f32x8 zero8() {
  f32x8 z;
#pragma unroll
  for (int i = 0; i < 8; ++i) z[i] = 0.f;
  return z;
}

__device__ __forceinline__ bf16x16 make_frag16(const __bf16* p0, const __bf16* p1) {
  bf16x8 lo = *(const bf16x8*)p0;
  bf16x8 hi = *(const bf16x8*)p1;
  bf16x16 r;
#pragma unroll
  for (int i = 0; i < 8; ++i) { r[i] = lo[i]; r[i + 8] = hi[i]; }
  return r;
}

// A fragment (16x32 bf16): lanes 0-15 row m hold k {0..7,16..23}; lanes 16-31 hold k {8..15,24..31}
__device__ __forceinline__ bf16x16 load_afrag(const __bf16* base, int ld, int row0, int k0, int lane) {
  int m = lane & 15, ha = lane >> 4;
  const __bf16* p = base + (size_t)(row0 + m) * ld + k0 + ha * 8;
  return make_frag16(p, p + 16);
}

// B fragment (32x16 bf16): lane n holds k {0..7,16..23}, lane n+16 holds k {8..15,24..31}.
// Source must be "transposed" layout: rows = output columns, contiguous in k.
__device__ __forceinline__ bf16x16 load_bfrag(const __bf16* base, int ld, int col0, int k0, int lane) {
  int n = lane & 15, hb = lane >> 4;
  const __bf16* p = base + (size_t)(col0 + n) * ld + k0 + hb * 8;
  return make_frag16(p, p + 16);
}

__device__ __forceinline__ f32x8 wmma_bf16(bf16x16 a, bf16x16 b, f32x8 c) {
  return __builtin_amdgcn_wmma_f32_16x16x32_bf16(false, a, false, b, (short)0, c, false, false);
}

// ---------------------------------------------------------------- GEMM ----
// C[M,N] = A[M,K] @ Bt[N,K]^T (+bias[N]) (+resid[M,N]); outputs f32 and/or bf16.
// Register-blocked: each wave computes a 64M x 32N macro-tile (4 A-frags x 2
// B-frags -> 8 wmma per 6 fragment loads). blockIdx.x walks M (fast) so a
// 128-column weight panel stays L2-resident while all M rows stream past it;
// the bf16 activation matrix (<=16 MB) is itself L2-resident.
__global__ __launch_bounds__(128) void gemm_bf16_kernel(
    const __bf16* __restrict__ A, const __bf16* __restrict__ Bt,
    const float* __restrict__ bias, const float* __restrict__ resid,
    float* __restrict__ Cf, __bf16* __restrict__ Cb, int M, int N, int Kd)
{
  int lane = threadIdx.x & 31;
  int wave = threadIdx.x >> 5;
  int tm0 = blockIdx.x * 4;            // 4 M-tiles (64 rows) per wave
  int tn0 = blockIdx.y * 8 + wave * 2; // 2 N-tiles (32 cols) per wave
  if (tn0 * 16 >= N) return;
  const __bf16* brow[2];
  const __bf16* arow[4];
#pragma unroll
  for (int j = 0; j < 2; ++j) {
    int nlc = (tn0 + j) * 16 + (lane & 15); if (nlc >= N) nlc = N - 1;
    brow[j] = Bt + (size_t)nlc * Kd + (lane >> 4) * 8;
  }
#pragma unroll
  for (int i = 0; i < 4; ++i) {
    int ml = (tm0 + i) * 16 + (lane & 15); if (ml >= M) ml = M - 1;
    arow[i] = A + (size_t)ml * Kd + (lane >> 4) * 8;
  }
  f32x8 acc[4][2];
#pragma unroll
  for (int i = 0; i < 4; ++i) { acc[i][0] = zero8(); acc[i][1] = zero8(); }

  for (int k0 = 0; k0 < Kd; k0 += 32) {
    __builtin_prefetch(brow[0] + k0 + 512, 0, 1);
    __builtin_prefetch(brow[1] + k0 + 512, 0, 1);
    bf16x16 b0 = make_frag16(brow[0] + k0, brow[0] + k0 + 16);
    bf16x16 b1 = make_frag16(brow[1] + k0, brow[1] + k0 + 16);
#pragma unroll
    for (int i = 0; i < 4; ++i) {
      __builtin_prefetch(arow[i] + k0 + 512, 0, 1);
      bf16x16 a = make_frag16(arow[i] + k0, arow[i] + k0 + 16);
      acc[i][0] = wmma_bf16(a, b0, acc[i][0]);
      acc[i][1] = wmma_bf16(a, b1, acc[i][1]);
    }
  }

  int hi = lane >> 4, nl = lane & 15;
#pragma unroll
  for (int j = 0; j < 2; ++j) {
    int col = (tn0 + j) * 16 + nl;
    if (col >= N) continue;
    float bv = bias ? bias[col] : 0.f;
#pragma unroll
    for (int i = 0; i < 4; ++i) {
#pragma unroll
      for (int r = 0; r < 8; ++r) {
        int row = (tm0 + i) * 16 + r + 8 * hi;
        if (row >= M) continue;
        float v = acc[i][j][r] + bv;
        if (resid) v += resid[(size_t)row * N + col];
        if (Cf) Cf[(size_t)row * N + col] = v;
        if (Cb) Cb[(size_t)row * N + col] = (__bf16)v;
      }
    }
  }
}

// Fused SwiGLU: out_bf = silu(A@W1t) * (A@W3t); 64M x 16N per wave against
// both weight matrices (4 A-frags x {b1,b3} -> 8 wmma per 6 fragment loads).
__global__ __launch_bounds__(128) void gemm_swiglu_kernel(
    const __bf16* __restrict__ A, const __bf16* __restrict__ B1t,
    const __bf16* __restrict__ B3t, __bf16* __restrict__ Cb, int M, int N, int Kd)
{
  int lane = threadIdx.x & 31;
  int wave = threadIdx.x >> 5;
  int tm0 = blockIdx.x * 4;
  int tn = blockIdx.y * 4 + wave;
  if (tn * 16 >= N) return;
  int nlc = tn * 16 + (lane & 15); if (nlc >= N) nlc = N - 1;
  const __bf16* b1row = B1t + (size_t)nlc * Kd + (lane >> 4) * 8;
  const __bf16* b3row = B3t + (size_t)nlc * Kd + (lane >> 4) * 8;
  const __bf16* arow[4];
#pragma unroll
  for (int i = 0; i < 4; ++i) {
    int ml = (tm0 + i) * 16 + (lane & 15); if (ml >= M) ml = M - 1;
    arow[i] = A + (size_t)ml * Kd + (lane >> 4) * 8;
  }
  f32x8 accg[4], accu[4];
#pragma unroll
  for (int i = 0; i < 4; ++i) { accg[i] = zero8(); accu[i] = zero8(); }

  for (int k0 = 0; k0 < Kd; k0 += 32) {
    __builtin_prefetch(b1row + k0 + 512, 0, 1);
    __builtin_prefetch(b3row + k0 + 512, 0, 1);
    bf16x16 b1 = make_frag16(b1row + k0, b1row + k0 + 16);
    bf16x16 b3 = make_frag16(b3row + k0, b3row + k0 + 16);
#pragma unroll
    for (int i = 0; i < 4; ++i) {
      __builtin_prefetch(arow[i] + k0 + 512, 0, 1);
      bf16x16 a = make_frag16(arow[i] + k0, arow[i] + k0 + 16);
      accg[i] = wmma_bf16(a, b1, accg[i]);
      accu[i] = wmma_bf16(a, b3, accu[i]);
    }
  }

  int hi = lane >> 4, nl = lane & 15;
  int col = tn * 16 + nl;
  if (col >= N) return;
#pragma unroll
  for (int i = 0; i < 4; ++i) {
#pragma unroll
    for (int r = 0; r < 8; ++r) {
      int row = (tm0 + i) * 16 + r + 8 * hi;
      if (row >= M) continue;
      float g = accg[i][r];
      float sg = g / (1.f + __expf(-g));
      Cb[(size_t)row * N + col] = (__bf16)(sg * accu[i][r]);
    }
  }
}

// ---------------------------------------------------------- Attention ----
// One wave per (b, h, 16-query tile). Flash-style over 32-key blocks, WMMA for
// QK^T and PV. Optional fused prompt attention (separate softmax, tanh gate).
__global__ __launch_bounds__(32) void attn_kernel(
    const __bf16* __restrict__ Q,    // [B][Hn][Sq][HD]
    const __bf16* __restrict__ K,    // [B][KVH][Skv][HD]
    const __bf16* __restrict__ Vt,   // [B][KVH][HD][Skv]
    const __bf16* __restrict__ AK,   // [B][KVH][16][HD] or null
    const __bf16* __restrict__ AVt,  // [B][KVH][HD][32] or null
    const float* __restrict__ gate,  // [Hn] or null
    __bf16* __restrict__ O,          // [B][Sq][Hn*HD]
    int Sq, int Skv, int KVH, int nrep, int causal, int Hn)
{
  const int lane = threadIdx.x;
  const int qt = blockIdx.x, h = blockIdx.y, b = blockIdx.z;
  const int kvh = h / nrep;
  const float scale = 0.088388347648318447f;  // 1/sqrt(128)
  const int hi = lane >> 4, nl = lane & 15;

  const __bf16* Qb = Q + ((size_t)(b * Hn + h) * Sq) * HD;
  const __bf16* Kb = K + ((size_t)(b * KVH + kvh) * Skv) * HD;
  const __bf16* Vb = Vt + ((size_t)(b * KVH + kvh) * HD) * Skv;

  __shared__ __bf16 p_lds[16 * 32];

  bf16x16 qf[4];
#pragma unroll
  for (int ks = 0; ks < 4; ++ks) qf[ks] = load_afrag(Qb, HD, qt * 16, ks * 32, lane);

  f32x8 Oacc[8];
#pragma unroll
  for (int dt = 0; dt < 8; ++dt) Oacc[dt] = zero8();
  float mrow[8], lrow[8];
#pragma unroll
  for (int r = 0; r < 8; ++r) { mrow[r] = -3.0e38f; lrow[r] = 0.f; }

  const int limit = causal ? (qt * 16 + 16) : Skv;
  const int nblk = (limit + 31) / 32;

  for (int jb = 0; jb < nblk; ++jb) {
    const int kbase = jb * 32;
    f32x8 s0 = zero8(), s1 = zero8();
#pragma unroll
    for (int ks = 0; ks < 4; ++ks) {
      bf16x16 kf0 = load_bfrag(Kb, HD, kbase, ks * 32, lane);
      bf16x16 kf1 = load_bfrag(Kb, HD, kbase + 16, ks * 32, lane);
      s0 = wmma_bf16(qf[ks], kf0, s0);
      s1 = wmma_bf16(qf[ks], kf1, s1);
    }
    float corr[8];
#pragma unroll
    for (int r = 0; r < 8; ++r) {
      const int qidx = qt * 16 + r + 8 * hi;
      const int k0i = kbase + nl, k1i = kbase + 16 + nl;
      float a = s0[r] * scale, c2 = s1[r] * scale;
      if ((causal && k0i > qidx) || k0i >= Skv) a = -3.0e38f;
      if ((causal && k1i > qidx) || k1i >= Skv) c2 = -3.0e38f;
      float mx = fmaxf(a, c2);
      mx = fmaxf(mx, __shfl_xor(mx, 1));
      mx = fmaxf(mx, __shfl_xor(mx, 2));
      mx = fmaxf(mx, __shfl_xor(mx, 4));
      mx = fmaxf(mx, __shfl_xor(mx, 8));
      float mnew = fmaxf(mrow[r], mx);
      float c = __expf(mrow[r] - mnew);
      float p0 = __expf(a - mnew);
      float p1 = __expf(c2 - mnew);
      float ts = p0 + p1;
      ts += __shfl_xor(ts, 1);
      ts += __shfl_xor(ts, 2);
      ts += __shfl_xor(ts, 4);
      ts += __shfl_xor(ts, 8);
      lrow[r] = lrow[r] * c + ts;
      mrow[r] = mnew;
      corr[r] = c;
      s0[r] = p0; s1[r] = p1;
    }
#pragma unroll
    for (int dt = 0; dt < 8; ++dt)
#pragma unroll
      for (int r = 0; r < 8; ++r) Oacc[dt][r] *= corr[r];
    __syncthreads();
#pragma unroll
    for (int r = 0; r < 8; ++r) {
      int m = r + 8 * hi;
      p_lds[m * 32 + nl]      = (__bf16)s0[r];
      p_lds[m * 32 + 16 + nl] = (__bf16)s1[r];
    }
    __syncthreads();
    bf16x16 pf;
    {
      int m = lane & 15, ha = lane >> 4;
      const __bf16* pr = &p_lds[m * 32 + ha * 8];
      pf = make_frag16(pr, pr + 16);
    }
#pragma unroll
    for (int dt = 0; dt < 8; ++dt) {
      bf16x16 vf = load_bfrag(Vb, Skv, dt * 16, kbase, lane);
      Oacc[dt] = wmma_bf16(pf, vf, Oacc[dt]);
    }
  }

  float invl[8];
#pragma unroll
  for (int r = 0; r < 8; ++r) invl[r] = 1.f / lrow[r];
#pragma unroll
  for (int dt = 0; dt < 8; ++dt)
#pragma unroll
    for (int r = 0; r < 8; ++r) Oacc[dt][r] *= invl[r];

  if (AK) {  // fused prompt attention: keys 0..AL-1, separate softmax, tanh gate
    const float g = tanhf(gate[h]);
    const __bf16* AKb = AK + ((size_t)(b * KVH + kvh) * 16) * HD;
    const __bf16* AVb = AVt + ((size_t)(b * KVH + kvh) * HD) * 32;
    f32x8 pa = zero8();
#pragma unroll
    for (int ks = 0; ks < 4; ++ks) {
      bf16x16 kf = load_bfrag(AKb, HD, 0, ks * 32, lane);
      pa = wmma_bf16(qf[ks], kf, pa);
    }
    float pl[8];
#pragma unroll
    for (int r = 0; r < 8; ++r) {
      float a = (nl < ALn) ? pa[r] * scale : -3.0e38f;
      float mx = a;
      mx = fmaxf(mx, __shfl_xor(mx, 1));
      mx = fmaxf(mx, __shfl_xor(mx, 2));
      mx = fmaxf(mx, __shfl_xor(mx, 4));
      mx = fmaxf(mx, __shfl_xor(mx, 8));
      float p0 = __expf(a - mx);
      float ts = p0;
      ts += __shfl_xor(ts, 1);
      ts += __shfl_xor(ts, 2);
      ts += __shfl_xor(ts, 4);
      ts += __shfl_xor(ts, 8);
      pl[r] = ts;
      pa[r] = p0;
    }
    __syncthreads();
#pragma unroll
    for (int r = 0; r < 8; ++r) {
      int m = r + 8 * hi;
      p_lds[m * 32 + nl]      = (__bf16)pa[r];
      p_lds[m * 32 + 16 + nl] = (__bf16)0.f;
    }
    __syncthreads();
    bf16x16 pf;
    {
      int m = lane & 15, ha = lane >> 4;
      const __bf16* pr = &p_lds[m * 32 + ha * 8];
      pf = make_frag16(pr, pr + 16);
    }
#pragma unroll
    for (int dt = 0; dt < 8; ++dt) {
      f32x8 t = zero8();
      bf16x16 vf = load_bfrag(AVb, 32, dt * 16, 0, lane);
      t = wmma_bf16(pf, vf, t);
#pragma unroll
      for (int r = 0; r < 8; ++r) Oacc[dt][r] += g * t[r] / pl[r];
    }
  }

#pragma unroll
  for (int dt = 0; dt < 8; ++dt)
#pragma unroll
    for (int r = 0; r < 8; ++r) {
      int srow = qt * 16 + r + 8 * hi;
      O[((size_t)b * Sq + srow) * ((size_t)Hn * HD) + (size_t)h * HD + dt * 16 + nl] =
          (__bf16)Oacc[dt][r];
    }
}

// -------------------------------------------------- data-prep kernels ----
// W[K][N] f32 -> Wt[N][K] bf16 (K,N multiples of 32)
__global__ void transpose_bf16_kernel(const float* __restrict__ in, __bf16* __restrict__ out,
                                      int K_, int N_) {
  __shared__ float tile[32][33];
  int n0 = blockIdx.x * 32, k0 = blockIdx.y * 32;
  int tx = threadIdx.x, ty = threadIdx.y;  // 32 x 8
#pragma unroll
  for (int i = 0; i < 32; i += 8)
    tile[ty + i][tx] = in[(size_t)(k0 + ty + i) * N_ + n0 + tx];
  __syncthreads();
#pragma unroll
  for (int i = 0; i < 32; i += 8)
    out[(size_t)(n0 + ty + i) * K_ + k0 + tx] = (__bf16)tile[tx][ty + i];
}

__global__ void rmsnorm_bf16_kernel(const float* __restrict__ x, const float* __restrict__ w,
                                    __bf16* __restrict__ out, int Dn) {
  __shared__ float red[256];
  int row = blockIdx.x, t = threadIdx.x;
  const float* xr = x + (size_t)row * Dn;
  float ss = 0.f;
  for (int i = t; i < Dn; i += 256) { float v = xr[i]; ss += v * v; }
  red[t] = ss; __syncthreads();
  for (int o = 128; o > 0; o >>= 1) { if (t < o) red[t] += red[t + o]; __syncthreads(); }
  float inv = rsqrtf(red[0] / Dn + 1e-5f);
  for (int i = t; i < Dn; i += 256) out[(size_t)row * Dn + i] = (__bf16)(xr[i] * inv * w[i]);
}

__global__ void layernorm_bf16_kernel(const float* __restrict__ x, const float* __restrict__ a,
                                      const float* __restrict__ bvec, __bf16* __restrict__ out, int Dn) {
  __shared__ float rs[256];
  __shared__ float rq[256];
  int row = blockIdx.x, t = threadIdx.x;
  const float* xr = x + (size_t)row * Dn;
  float s = 0.f, q = 0.f;
  for (int i = t; i < Dn; i += 256) { float v = xr[i]; s += v; q += v * v; }
  rs[t] = s; rq[t] = q; __syncthreads();
  for (int o = 128; o > 0; o >>= 1) { if (t < o) { rs[t] += rs[t + o]; rq[t] += rq[t + o]; } __syncthreads(); }
  float mean = rs[0] / Dn;
  float var = (rq[0] - Dn * mean * mean) / (Dn - 1);  // ddof=1
  float inv = 1.f / (sqrtf(var) + 1e-6f);
  for (int i = t; i < Dn; i += 256)
    out[(size_t)row * Dn + i] = (__bf16)(a[i] * (xr[i] - mean) * inv + bvec[i]);
}

__global__ void f32_to_bf16_kernel(const float* __restrict__ in, __bf16* __restrict__ out, size_t n) {
  size_t i = (size_t)blockIdx.x * blockDim.x + threadIdx.x;
  if (i < n) out[i] = (__bf16)in[i];
}

// in [B][S][Heads*HD] f32 (GEMM output) -> out [B][Heads][S][HD] bf16 with RoPE
__global__ void rope_reorder_kernel(const float* __restrict__ in, const float* __restrict__ cb,
                                    const float* __restrict__ sb, __bf16* __restrict__ out,
                                    int S_, int Heads) {
  size_t tid = (size_t)blockIdx.x * blockDim.x + threadIdx.x;
  size_t total = (size_t)Bb * Heads * S_ * (HD / 2);
  if (tid >= total) return;
  int i = tid & 63;
  int s = (int)((tid >> 6) % S_);
  int hh = (int)((tid / ((size_t)64 * S_)) % Heads);
  int b = (int)(tid / ((size_t)64 * S_ * Heads));
  const float* ip = in + (size_t)(b * S_ + s) * (Heads * HD) + hh * HD + 2 * i;
  float t0 = ip[0], t1 = ip[1];
  float c = cb[s * 64 + i], sn = sb[s * 64 + i];
  __bf16* op = out + ((size_t)(b * Heads + hh) * S_ + s) * HD + 2 * i;
  op[0] = (__bf16)(t0 * c - t1 * sn);
  op[1] = (__bf16)(t0 * sn + t1 * c);
}

// in [B][S][Heads*HD] f32 -> out [B][Heads][S][HD] bf16 (no rope)
__global__ void reorder_heads_kernel(const float* __restrict__ in, __bf16* __restrict__ out,
                                     int S_, int Heads) {
  size_t tid = (size_t)blockIdx.x * blockDim.x + threadIdx.x;
  size_t total = (size_t)Bb * Heads * S_ * HD;
  if (tid >= total) return;
  int d = tid & 127;
  int s = (int)((tid >> 7) % S_);
  int hh = (int)((tid / ((size_t)HD * S_)) % Heads);
  int b = (int)(tid / ((size_t)HD * S_ * Heads));
  out[((size_t)(b * Heads + hh) * S_ + s) * HD + d] =
      (__bf16)in[(size_t)(b * S_ + s) * (Heads * HD) + hh * HD + d];
}

// in [B][S][Heads*HD] f32 -> out [B][Heads][HD][S] bf16 (transposed V)
__global__ void v_transpose_kernel(const float* __restrict__ in, __bf16* __restrict__ out,
                                   int S_, int Heads) {
  size_t tid = (size_t)blockIdx.x * blockDim.x + threadIdx.x;
  size_t total = (size_t)Bb * Heads * HD * S_;
  if (tid >= total) return;
  int s = (int)(tid % S_);
  int d = (int)((tid / S_) % HD);
  int hh = (int)((tid / ((size_t)S_ * HD)) % Heads);
  int b = (int)(tid / ((size_t)S_ * HD * Heads));
  out[((size_t)(b * Heads + hh) * HD + d) * S_ + s] =
      (__bf16)in[(size_t)(b * S_ + s) * (Heads * HD) + hh * HD + d];
}

// ak f32 [B][AL][KVh*HD] -> AK bf16 [B][KVh][16][HD], zero padded keys
__global__ void pad_prompt_k_kernel(const float* __restrict__ ak, __bf16* __restrict__ AK) {
  int tid = blockIdx.x * blockDim.x + threadIdx.x;
  if (tid >= Bb * KVh * 16 * HD) return;
  int d = tid & 127;
  int key = (tid >> 7) & 15;
  int kv = (tid >> 11) & 7;
  int b = tid >> 14;
  float v = (key < ALn) ? ak[(size_t)(b * ALn + key) * (KVh * HD) + kv * HD + d] : 0.f;
  AK[tid] = (__bf16)v;
}

// av f32 [B][AL][KVh*HD] -> AVt bf16 [B][KVh][HD][32], zero padded keys
__global__ void pad_prompt_v_kernel(const float* __restrict__ av, __bf16* __restrict__ AVt) {
  int tid = blockIdx.x * blockDim.x + threadIdx.x;
  if (tid >= Bb * KVh * HD * 32) return;
  int key = tid & 31;
  int d = (tid >> 5) & 127;
  int kv = (tid >> 12) & 7;
  int b = tid >> 15;
  float v = (key < ALn) ? av[(size_t)(b * ALn + key) * (KVh * HD) + kv * HD + d] : 0.f;
  AVt[tid] = (__bf16)v;
}

// ----------------------------------------------------------- launcher ----
static inline void launch_gemm(const __bf16* A, const __bf16* Bt, const float* bias,
                               const float* resid, float* Cf, __bf16* Cb,
                               int M, int N, int Kd, hipStream_t s) {
  dim3 g((M + 63) / 64, (N + 127) / 128);  // x walks M (fast) for L2 weight reuse
  gemm_bf16_kernel<<<g, 128, 0, s>>>(A, Bt, bias, resid, Cf, Cb, M, N, Kd);
}

extern "C" void kernel_launch(void* const* d_in, const int* in_sizes, int n_in,
                              void* d_out, int out_size, void* d_ws, size_t ws_size,
                              hipStream_t stream) {
  (void)in_sizes; (void)n_in; (void)out_size; (void)ws_size;
  const float* x        = (const float*)d_in[0];
  const float* fcos     = (const float*)d_in[2];
  const float* fsin     = (const float*)d_in[3];
  const float* prompt   = (const float*)d_in[5];
  const float* regtok   = (const float*)d_in[6];
  const float* wq       = (const float*)d_in[7];
  const float* wk       = (const float*)d_in[8];
  const float* wv       = (const float*)d_in[9];
  const float* wo       = (const float*)d_in[10];
  const float* gate     = (const float*)d_in[11];
  const float* w1       = (const float*)d_in[12];
  const float* w2       = (const float*)d_in[13];
  const float* w3       = (const float*)d_in[14];
  const float* attn_nw  = (const float*)d_in[15];
  const float* ffn_nw   = (const float*)d_in[16];
  const float* ln_a     = (const float*)d_in[17];
  const float* ln_b     = (const float*)d_in[18];
  const float* mq_w     = (const float*)d_in[19];
  const float* mq_b     = (const float*)d_in[20];
  const float* mk_w     = (const float*)d_in[21];
  const float* mk_b     = (const float*)d_in[22];
  const float* mv_w     = (const float*)d_in[23];
  const float* mv_b     = (const float*)d_in[24];
  const float* mo_w     = (const float*)d_in[25];
  const float* mo_b     = (const float*)d_in[26];

  char* wsp = (char*)d_ws;
  size_t off = 0;
  auto alloc = [&](size_t bytes) -> void* {
    void* p = wsp + off;
    off += (bytes + 255) & ~(size_t)255;
    return p;
  };

  const int M = Bb * Ss;          // 2048 tokens
  const int MR = Bb * RLn;        // 256 register tokens
  const int MP = Bb * ALn;        // 40 prompt tokens

  __bf16* wqT = (__bf16*)alloc((size_t)Dd * Dd * 2);
  __bf16* wkT = (__bf16*)alloc((size_t)(KVh * HD) * Dd * 2);
  __bf16* wvT = (__bf16*)alloc((size_t)(KVh * HD) * Dd * 2);
  __bf16* woT = (__bf16*)alloc((size_t)Dd * Dd * 2);
  __bf16* w1T = (__bf16*)alloc((size_t)HID * Dd * 2);
  __bf16* w3T = (__bf16*)alloc((size_t)HID * Dd * 2);
  __bf16* w2T = (__bf16*)alloc((size_t)Dd * HID * 2);
  __bf16* mqT = (__bf16*)alloc((size_t)Dd * Dd * 2);
  __bf16* mkT = (__bf16*)alloc((size_t)Dd * Dd * 2);
  __bf16* mvT = (__bf16*)alloc((size_t)Dd * Dd * 2);
  __bf16* moT = (__bf16*)alloc((size_t)Dd * Dd * 2);

  __bf16* xn_bf  = (__bf16*)alloc((size_t)M * Dd * 2);
  float*  xq_f   = (float*)alloc((size_t)M * Dd * 4);
  float*  xk_f   = (float*)alloc((size_t)M * (KVh * HD) * 4);
  float*  xv_f   = (float*)alloc((size_t)M * (KVh * HD) * 4);
  __bf16* Qbf    = (__bf16*)alloc((size_t)M * Dd * 2);
  __bf16* Kbf    = (__bf16*)alloc((size_t)Bb * KVh * Ss * HD * 2);
  __bf16* Vtbf   = (__bf16*)alloc((size_t)Bb * KVh * HD * Ss * 2);
  __bf16* prm_bf = (__bf16*)alloc((size_t)MP * Dd * 2);
  float*  ak_f   = (float*)alloc((size_t)MP * (KVh * HD) * 4);
  float*  av_f   = (float*)alloc((size_t)MP * (KVh * HD) * 4);
  __bf16* AKbf   = (__bf16*)alloc((size_t)Bb * KVh * 16 * HD * 2);
  __bf16* AVtbf  = (__bf16*)alloc((size_t)Bb * KVh * HD * 32 * 2);
  __bf16* o_bf   = (__bf16*)alloc((size_t)M * Dd * 2);
  float*  h_f    = (float*)alloc((size_t)M * Dd * 4);
  __bf16* hn_bf  = (__bf16*)alloc((size_t)M * Dd * 2);
  __bf16* s_bf   = (__bf16*)alloc((size_t)M * HID * 2);
  __bf16* rn_bf  = (__bf16*)alloc((size_t)MR * Dd * 2);
  float*  rq_f   = (float*)alloc((size_t)MR * Dd * 4);
  float*  rk_f   = (float*)alloc((size_t)MR * Dd * 4);
  float*  rv_f   = (float*)alloc((size_t)MR * Dd * 4);
  __bf16* Qr     = (__bf16*)alloc((size_t)MR * Dd * 2);
  __bf16* Kr     = (__bf16*)alloc((size_t)MR * Dd * 2);
  __bf16* Vtr    = (__bf16*)alloc((size_t)MR * Dd * 2);
  __bf16* ry_bf  = (__bf16*)alloc((size_t)MR * Dd * 2);

  float* out_main = (float*)d_out;
  float* out_reg  = out_main + (size_t)M * Dd;

  dim3 tb(32, 8);
  // weight transposes (f32 -> bf16, N x K)
  transpose_bf16_kernel<<<dim3(Dd / 32, Dd / 32), tb, 0, stream>>>(wq, wqT, Dd, Dd);
  transpose_bf16_kernel<<<dim3((KVh * HD) / 32, Dd / 32), tb, 0, stream>>>(wk, wkT, Dd, KVh * HD);
  transpose_bf16_kernel<<<dim3((KVh * HD) / 32, Dd / 32), tb, 0, stream>>>(wv, wvT, Dd, KVh * HD);
  transpose_bf16_kernel<<<dim3(Dd / 32, Dd / 32), tb, 0, stream>>>(wo, woT, Dd, Dd);
  transpose_bf16_kernel<<<dim3(HID / 32, Dd / 32), tb, 0, stream>>>(w1, w1T, Dd, HID);
  transpose_bf16_kernel<<<dim3(HID / 32, Dd / 32), tb, 0, stream>>>(w3, w3T, Dd, HID);
  transpose_bf16_kernel<<<dim3(Dd / 32, HID / 32), tb, 0, stream>>>(w2, w2T, HID, Dd);
  transpose_bf16_kernel<<<dim3(Dd / 32, Dd / 32), tb, 0, stream>>>(mq_w, mqT, Dd, Dd);
  transpose_bf16_kernel<<<dim3(Dd / 32, Dd / 32), tb, 0, stream>>>(mk_w, mkT, Dd, Dd);
  transpose_bf16_kernel<<<dim3(Dd / 32, Dd / 32), tb, 0, stream>>>(mv_w, mvT, Dd, Dd);
  transpose_bf16_kernel<<<dim3(Dd / 32, Dd / 32), tb, 0, stream>>>(mo_w, moT, Dd, Dd);

  // attention branch
  rmsnorm_bf16_kernel<<<M, 256, 0, stream>>>(x, attn_nw, xn_bf, Dd);
  launch_gemm(xn_bf, wqT, nullptr, nullptr, xq_f, nullptr, M, Dd, Dd, stream);
  launch_gemm(xn_bf, wkT, nullptr, nullptr, xk_f, nullptr, M, KVh * HD, Dd, stream);
  launch_gemm(xn_bf, wvT, nullptr, nullptr, xv_f, nullptr, M, KVh * HD, Dd, stream);

  {
    size_t n = (size_t)MP * Dd;
    f32_to_bf16_kernel<<<(unsigned)((n + 255) / 256), 256, 0, stream>>>(prompt, prm_bf, n);
  }
  launch_gemm(prm_bf, wkT, nullptr, nullptr, ak_f, nullptr, MP, KVh * HD, Dd, stream);
  launch_gemm(prm_bf, wvT, nullptr, nullptr, av_f, nullptr, MP, KVh * HD, Dd, stream);

  {
    size_t nq = (size_t)Bb * Hh * Ss * 64;
    rope_reorder_kernel<<<(unsigned)((nq + 255) / 256), 256, 0, stream>>>(xq_f, fcos, fsin, Qbf, Ss, Hh);
    size_t nk = (size_t)Bb * KVh * Ss * 64;
    rope_reorder_kernel<<<(unsigned)((nk + 255) / 256), 256, 0, stream>>>(xk_f, fcos, fsin, Kbf, Ss, KVh);
    size_t nv = (size_t)Bb * KVh * HD * Ss;
    v_transpose_kernel<<<(unsigned)((nv + 255) / 256), 256, 0, stream>>>(xv_f, Vtbf, Ss, KVh);
  }
  pad_prompt_k_kernel<<<(Bb * KVh * 16 * HD + 255) / 256, 256, 0, stream>>>(ak_f, AKbf);
  pad_prompt_v_kernel<<<(Bb * KVh * HD * 32 + 255) / 256, 256, 0, stream>>>(av_f, AVtbf);

  attn_kernel<<<dim3(Ss / 16, Hh, Bb), 32, 0, stream>>>(
      Qbf, Kbf, Vtbf, AKbf, AVtbf, gate, o_bf, Ss, Ss, KVh, Hh / KVh, 1, Hh);

  launch_gemm(o_bf, woT, nullptr, x, h_f, nullptr, M, Dd, Dd, stream);

  // FFN branch
  rmsnorm_bf16_kernel<<<M, 256, 0, stream>>>(h_f, ffn_nw, hn_bf, Dd);
  {
    dim3 g((M + 63) / 64, (HID + 63) / 64);
    gemm_swiglu_kernel<<<g, 128, 0, stream>>>(hn_bf, w1T, w3T, s_bf, M, HID, Dd);
  }
  launch_gemm(s_bf, w2T, nullptr, h_f, out_main, nullptr, M, Dd, HID, stream);

  // register-token branch
  layernorm_bf16_kernel<<<MR, 256, 0, stream>>>(regtok, ln_a, ln_b, rn_bf, Dd);
  launch_gemm(rn_bf, mqT, mq_b, nullptr, rq_f, nullptr, MR, Dd, Dd, stream);
  launch_gemm(rn_bf, mkT, mk_b, nullptr, rk_f, nullptr, MR, Dd, Dd, stream);
  launch_gemm(rn_bf, mvT, mv_b, nullptr, rv_f, nullptr, MR, Dd, Dd, stream);
  {
    size_t n = (size_t)Bb * Hh * RLn * HD;
    reorder_heads_kernel<<<(unsigned)((n + 255) / 256), 256, 0, stream>>>(rq_f, Qr, RLn, Hh);
    reorder_heads_kernel<<<(unsigned)((n + 255) / 256), 256, 0, stream>>>(rk_f, Kr, RLn, Hh);
    v_transpose_kernel<<<(unsigned)((n + 255) / 256), 256, 0, stream>>>(rv_f, Vtr, RLn, Hh);
  }
  attn_kernel<<<dim3(RLn / 16, Hh, Bb), 32, 0, stream>>>(
      Qr, Kr, Vtr, nullptr, nullptr, nullptr, ry_bf, RLn, RLn, Hh, 1, 0, Hh);

  launch_gemm(ry_bf, moT, mo_b, regtok, out_reg, nullptr, MR, Dd, Dd, stream);
}
